// SlidingWindowAttention_74500502716610
// MI455X (gfx1250) — matrix operand order repaired and checked by reading, not compiled, not run on hardware
//
#include <hip/hip_runtime.h>

#ifndef __has_builtin
#define __has_builtin(x) 0
#endif

// Problem constants (match reference)
static constexpr int kDim = 1024;
static constexpr int kSeq = 2048;
static constexpr int kB   = 2;
static constexpr int kH   = 16;
static constexpr int kHD  = 64;
static constexpr int kWin = 256;
static constexpr int kM   = kB * kSeq;  // 4096 rows in all projections

typedef __attribute__((ext_vector_type(16))) __bf16        v16bf;
typedef __attribute__((ext_vector_type(8)))  float         v8f;
typedef __attribute__((ext_vector_type(4)))  unsigned int  u32x4;
typedef __attribute__((ext_vector_type(4)))  float         f32x4;
typedef __attribute__((ext_vector_type(4)))  unsigned short u16x4;
typedef int b128_t __attribute__((vector_size(16)));  // matches builtin param type

union FragAB {
  v16bf v;
  u32x4 q[2];
  unsigned short us[16];
};

__device__ __forceinline__ unsigned short f2bf(float f) {
  unsigned int u = __float_as_uint(f);
  u += 0x7fffu + ((u >> 16) & 1u);   // round-to-nearest-even
  return (unsigned short)(u >> 16);
}

__device__ __forceinline__ u32x4 ld16(const unsigned short* p) {
  return *reinterpret_cast<const u32x4*>(p);
}

// ---------------------------------------------------------------------------
// Async global -> LDS copy (CDNA5 GLOBAL_LOAD_ASYNC_TO_LDS_B128, ASYNCcnt),
// with a synchronous ds_store fallback if the builtin is unavailable.
// Builtin signature (from compiler diagnostic):
//   void __builtin_amdgcn_global_load_async_to_lds_b128(
//       int __vector(4) AS1* gsrc, int __vector(4) AS3* ldst, imm int, imm int)
// ---------------------------------------------------------------------------
#if __has_builtin(__builtin_amdgcn_global_load_async_to_lds_b128)
#define ASYNC_COPY 1
__device__ __forceinline__ void async_cp16(const unsigned short* g, unsigned short* l) {
  __builtin_amdgcn_global_load_async_to_lds_b128(
      (__attribute__((address_space(1))) b128_t*)(g),
      (__attribute__((address_space(3))) b128_t*)(l), 0, 0);
}
#else
#define ASYNC_COPY 0
__device__ __forceinline__ void async_cp16(const unsigned short* g, unsigned short* l) {
  *reinterpret_cast<u32x4*>(l) = ld16(g);
}
#endif

#if ASYNC_COPY
#if __has_builtin(__builtin_amdgcn_s_wait_asynccnt)
#define WAIT_ASYNC(n) __builtin_amdgcn_s_wait_asynccnt(n)
#else
#define WAIT_ASYNC(n) asm volatile("s_wait_asynccnt %0" ::"n"(n) : "memory")
#endif
#else
#define WAIT_ASYNC(n) ((void)0)
#endif

// ---------------------------------------------------------------------------
// fp32 -> bf16 elementwise (vectorized x4)
// ---------------------------------------------------------------------------
__global__ __launch_bounds__(256) void cvt_f32_bf16(const float* __restrict__ in,
                                                    unsigned short* __restrict__ out,
                                                    int n4) {
  int i = blockIdx.x * blockDim.x + threadIdx.x;
  if (i < n4) {
    f32x4 x = reinterpret_cast<const f32x4*>(in)[i];
    u16x4 o;
    o.x = f2bf(x.x); o.y = f2bf(x.y); o.z = f2bf(x.z); o.w = f2bf(x.w);
    reinterpret_cast<u16x4*>(out)[i] = o;
  }
}

// ---------------------------------------------------------------------------
// fp32 [K,N] -> bf16 [N,K] (transpose + convert) for 1024x1024 weights
// ---------------------------------------------------------------------------
__global__ __launch_bounds__(256) void transpose_cvt(const float* __restrict__ W,
                                                     unsigned short* __restrict__ Wt) {
  __shared__ float tile[16][17];
  const int k = blockIdx.y * 16 + threadIdx.y;   // row of W
  const int n = blockIdx.x * 16 + threadIdx.x;   // col of W
  tile[threadIdx.y][threadIdx.x] = W[k * kDim + n];
  __syncthreads();
  const int on = blockIdx.x * 16 + threadIdx.y;  // row of Wt (= n)
  const int ok = blockIdx.y * 16 + threadIdx.x;  // col of Wt (= k)
  Wt[on * kDim + ok] = f2bf(tile[threadIdx.x][threadIdx.y]);
}

// ---------------------------------------------------------------------------
// GEMM: C[M=4096, N=1024] = A(bf16 [M,K] row-major) x B(given as Bt bf16
// [N,K] row-major) + bias(f32).
// 256 thr / 8 waves; block tile 128x128; wave tile 32x64 (8 WMMAs / K-step).
// Double-buffered async global->LDS staging (4 async b128 per thread/stage).
// MODE 0: bf16 -> [B,H,S,HD]   (Q, K)
// MODE 1: bf16 -> [B,H,HD,S]   (V transposed)
// MODE 2: f32  -> [M,N]        (final output)
// ---------------------------------------------------------------------------
static constexpr int kLdsPad = 40;  // 32 elements + 8 pad (16B-aligned rows)

template <int MODE>
__global__ __launch_bounds__(256) void gemm_bf16(const unsigned short* __restrict__ A,
                                                 const unsigned short* __restrict__ Bt,
                                                 const float* __restrict__ bias,
                                                 void* __restrict__ outp) {
  __shared__ unsigned short ldsA[2][128 * kLdsPad];
  __shared__ unsigned short ldsB[2][128 * kLdsPad];

  const int tid  = threadIdx.x;
  const int lane = tid & 31;
  const int wave = tid >> 5;
  const int hf   = lane >> 4;
  const int l16  = lane & 15;
  const int waveM = wave & 3;   // 4 waves along M
  const int waveN = wave >> 2;  // 2 waves along N
  const int blockRow = blockIdx.x * 128;
  const int blockCol = blockIdx.y * 128;

  const v8f vzero = {0.f, 0.f, 0.f, 0.f, 0.f, 0.f, 0.f, 0.f};
  v8f acc[2][4];
#pragma unroll
  for (int mt = 0; mt < 2; ++mt)
#pragma unroll
    for (int nt = 0; nt < 4; ++nt) acc[mt][nt] = vzero;

  // Stage one 128x32 A tile + 128x32 B tile into LDS buffer `buf`.
  auto stage = [&](int buf, int kk) {
#pragma unroll
    for (int c = 0; c < 2; ++c) {
      const int ch  = tid + c * 256;       // 512 chunks of 8 elements
      const int row = ch >> 2;
      const int col = (ch & 3) * 8;
      async_cp16(A + (size_t)(blockRow + row) * kDim + kk + col,
                 &ldsA[buf][row * kLdsPad + col]);
    }
#pragma unroll
    for (int c = 0; c < 2; ++c) {
      const int ch  = tid + c * 256;
      const int row = ch >> 2;
      const int col = (ch & 3) * 8;
      async_cp16(Bt + (size_t)(blockCol + row) * kDim + kk + col,
                 &ldsB[buf][row * kLdsPad + col]);
    }
  };

  stage(0, 0);
  for (int kk = 0; kk < kDim; kk += 32) {
    const int cur = (kk >> 5) & 1;
    if (kk + 32 < kDim) {
      stage(cur ^ 1, kk + 32);
      WAIT_ASYNC(4);   // previous batch (4 per wave) retired, next in flight
    } else {
      WAIT_ASYNC(0);
    }
    __syncthreads();

    FragAB bfr[4];
#pragma unroll
    for (int nt = 0; nt < 4; ++nt) {
      const unsigned short* bp =
          &ldsB[cur][(waveN * 64 + nt * 16 + l16) * kLdsPad + hf * 8];
      bfr[nt].q[0] = ld16(bp);
      bfr[nt].q[1] = ld16(bp + 16);
    }
#pragma unroll
    for (int mt = 0; mt < 2; ++mt) {
      FragAB afr;
      const unsigned short* ap = &ldsA[cur][(waveM * 32 + mt * 16 + l16) * kLdsPad];
      afr.q[0] = ld16(ap + hf * 8);
      afr.q[1] = ld16(ap + 16 + hf * 8);
#pragma unroll
      for (int nt = 0; nt < 4; ++nt)
        acc[mt][nt] = __builtin_amdgcn_wmma_f32_16x16x32_bf16(
            false, afr.v, false, bfr[nt].v, (short)0, acc[mt][nt], false, false);
    }
    __syncthreads();
  }

#pragma unroll
  for (int mt = 0; mt < 2; ++mt) {
#pragma unroll
    for (int nt = 0; nt < 4; ++nt) {
      const int n = blockCol + waveN * 64 + nt * 16 + l16;
      const float bval = bias[n];
#pragma unroll
      for (int r = 0; r < 8; ++r) {
        const int m = blockRow + waveM * 32 + mt * 16 + r + 8 * hf;
        const float val = acc[mt][nt][r] + bval;
        if (MODE == 0) {
          const int bb = m >> 11, s = m & (kSeq - 1);
          const int h = n >> 6, d = n & (kHD - 1);
          ((unsigned short*)outp)[(((size_t)(bb * kH + h) * kSeq + s) * kHD) + d] = f2bf(val);
        } else if (MODE == 1) {
          const int bb = m >> 11, s = m & (kSeq - 1);
          const int h = n >> 6, d = n & (kHD - 1);
          ((unsigned short*)outp)[(((size_t)(bb * kH + h) * kHD + d) * kSeq) + s] = f2bf(val);
        } else {
          ((float*)outp)[(size_t)m * kDim + n] = val;
        }
      }
    }
  }
}

// ---------------------------------------------------------------------------
// Sliding-window flash attention.  One wave (32 thr) per 16-query tile.
// Q,K: bf16 [B,H,S,HD].  Vt: bf16 [B,H,HD,S].  Output: bf16 [B*S, DIM].
// Per iteration: 32 keys -> 4 score WMMAs + 4 PV WMMAs.
// ---------------------------------------------------------------------------
__global__ __launch_bounds__(32) void attn_kernel(const unsigned short* __restrict__ Q,
                                                  const unsigned short* __restrict__ K,
                                                  const unsigned short* __restrict__ Vt,
                                                  unsigned short* __restrict__ O) {
  __shared__ unsigned short ldsP[16 * 32];  // P tile staging (C-frag -> A-frag)

  const int lane = threadIdx.x;
  const int hf   = lane >> 4;
  const int l16  = lane & 15;
  const int i0   = blockIdx.x * 16;
  const int h    = blockIdx.y;
  const int b    = blockIdx.z;

  const unsigned short* Qbh = Q  + (size_t)(b * kH + h) * kSeq * kHD;
  const unsigned short* Kbh = K  + (size_t)(b * kH + h) * kSeq * kHD;
  const unsigned short* Vbh = Vt + (size_t)(b * kH + h) * kHD * kSeq;

  // Q A-fragments for the two K=32 chunks of HD=64 (loaded once)
  FragAB qf[2];
  const unsigned short* qrow = Qbh + (size_t)(i0 + l16) * kHD;
#pragma unroll
  for (int c = 0; c < 2; ++c) {
    qf[c].q[0] = ld16(qrow + c * 32 + hf * 8);
    qf[c].q[1] = ld16(qrow + c * 32 + 16 + hf * 8);
  }

  const v8f vzero = {0.f, 0.f, 0.f, 0.f, 0.f, 0.f, 0.f, 0.f};
  v8f accO[4];
  accO[0] = vzero; accO[1] = vzero; accO[2] = vzero; accO[3] = vzero;
  float mrow[8], lrow[8];
#pragma unroll
  for (int r = 0; r < 8; ++r) { mrow[r] = -1e30f; lrow[r] = 0.f; }

  const int js = (i0 >= kWin) ? ((i0 - kWin) & ~31) : 0;
  for (int j0 = js; j0 < i0 + 16; j0 += 32) {
    // ---- scores: S[16q x 32k] as two 16x16 C-fragments
    v8f s0 = vzero, s1 = vzero;
#pragma unroll
    for (int c = 0; c < 2; ++c) {
      FragAB kf0, kf1;
      const unsigned short* kr0 = Kbh + (size_t)(j0 + l16) * kHD + c * 32 + hf * 8;
      const unsigned short* kr1 = kr0 + (size_t)16 * kHD;
      kf0.q[0] = ld16(kr0); kf0.q[1] = ld16(kr0 + 16);
      kf1.q[0] = ld16(kr1); kf1.q[1] = ld16(kr1 + 16);
      s0 = __builtin_amdgcn_wmma_f32_16x16x32_bf16(false, qf[c].v, false, kf0.v, (short)0, s0, false, false);
      s1 = __builtin_amdgcn_wmma_f32_16x16x32_bf16(false, qf[c].v, false, kf1.v, (short)0, s1, false, false);
    }

    // ---- scale + window mask + per-row block max
    float bmax[8];
#pragma unroll
    for (int r = 0; r < 8; ++r) {
      const int i = i0 + r + 8 * hf;
      const int ja = j0 + l16;
      const int jb = ja + 16;
      float va = s0[r] * 0.125f;
      float vb = s1[r] * 0.125f;
      va = ((ja <= i) && (ja + kWin > i)) ? va : -1e30f;
      vb = ((jb <= i) && (jb + kWin > i)) ? vb : -1e30f;
      s0[r] = va; s1[r] = vb;
      bmax[r] = fmaxf(va, vb);
    }
#pragma unroll
    for (int off = 1; off < 16; off <<= 1) {
#pragma unroll
      for (int r = 0; r < 8; ++r)
        bmax[r] = fmaxf(bmax[r], __shfl_xor(bmax[r], off, 32));
    }

    // ---- online softmax state
    float alpha[8], bsum[8];
#pragma unroll
    for (int r = 0; r < 8; ++r) {
      const float mnew = fmaxf(mrow[r], bmax[r]);
      alpha[r] = __expf(mrow[r] - mnew);
      mrow[r] = mnew;
      const float p0 = __expf(s0[r] - mnew);
      const float p1 = __expf(s1[r] - mnew);
      s0[r] = p0; s1[r] = p1;
      bsum[r] = p0 + p1;
    }
#pragma unroll
    for (int off = 1; off < 16; off <<= 1) {
#pragma unroll
      for (int r = 0; r < 8; ++r)
        bsum[r] += __shfl_xor(bsum[r], off, 32);
    }
#pragma unroll
    for (int r = 0; r < 8; ++r) lrow[r] = lrow[r] * alpha[r] + bsum[r];
#pragma unroll
    for (int nt = 0; nt < 4; ++nt)
#pragma unroll
      for (int r = 0; r < 8; ++r) accO[nt][r] *= alpha[r];

    // ---- transpose P through LDS: C-frag layout -> A-frag layout (bf16)
#pragma unroll
    for (int r = 0; r < 8; ++r) {
      const int m = r + 8 * hf;
      ldsP[m * 32 + l16]      = f2bf(s0[r]);
      ldsP[m * 32 + 16 + l16] = f2bf(s1[r]);
    }
    __syncthreads();  // single-wave WG: DS ops in-order; barrier is a NOP
    FragAB pf;
    pf.q[0] = ld16(&ldsP[l16 * 32 + hf * 8]);
    pf.q[1] = ld16(&ldsP[l16 * 32 + 16 + hf * 8]);

    // ---- accO += P[16x32] x V[32x64]
#pragma unroll
    for (int nt = 0; nt < 4; ++nt) {
      const unsigned short* vr = Vbh + (size_t)(nt * 16 + l16) * kSeq + j0 + hf * 8;
      FragAB vf;
      vf.q[0] = ld16(vr); vf.q[1] = ld16(vr + 16);
      accO[nt] = __builtin_amdgcn_wmma_f32_16x16x32_bf16(false, pf.v, false, vf.v, (short)0, accO[nt], false, false);
    }
    __syncthreads();
  }

  // ---- normalize + store to [B*S, DIM] bf16 for the output projection
#pragma unroll
  for (int nt = 0; nt < 4; ++nt) {
#pragma unroll
    for (int r = 0; r < 8; ++r) {
      const int m = r + 8 * hf;
      const int srow = i0 + m;
      const int d = nt * 16 + l16;
      const float val = accO[nt][r] / lrow[r];
      O[((size_t)(b * kSeq + srow)) * kDim + h * kHD + d] = f2bf(val);
    }
  }
}

// ---------------------------------------------------------------------------
// Launcher
// ---------------------------------------------------------------------------
extern "C" void kernel_launch(void* const* d_in, const int* in_sizes, int n_in,
                              void* d_out, int out_size, void* d_ws, size_t ws_size,
                              hipStream_t stream) {
  (void)in_sizes; (void)n_in; (void)out_size; (void)ws_size;
  const float* x  = (const float*)d_in[0];
  const float* Wq = (const float*)d_in[1];
  const float* bq = (const float*)d_in[2];
  const float* Wk = (const float*)d_in[3];
  const float* bk = (const float*)d_in[4];
  const float* Wv = (const float*)d_in[5];
  const float* bv = (const float*)d_in[6];
  const float* Wo = (const float*)d_in[7];
  const float* bo = (const float*)d_in[8];

  char* ws = (char*)d_ws;
  const size_t MB = 1ull << 20;
  unsigned short* xbf  = (unsigned short*)(ws + 0);        // 8 MB  [M,K] bf16
  unsigned short* WqT  = (unsigned short*)(ws + 8  * MB);  // 2 MB
  unsigned short* WkT  = (unsigned short*)(ws + 10 * MB);  // 2 MB
  unsigned short* WvT  = (unsigned short*)(ws + 12 * MB);  // 2 MB
  unsigned short* WoT  = (unsigned short*)(ws + 14 * MB);  // 2 MB
  unsigned short* Qb   = (unsigned short*)(ws + 16 * MB);  // 8 MB [B,H,S,HD]
  unsigned short* Kb   = (unsigned short*)(ws + 24 * MB);  // 8 MB [B,H,S,HD]
  unsigned short* Vtb  = (unsigned short*)(ws + 32 * MB);  // 8 MB [B,H,HD,S]
  unsigned short* AObf = (unsigned short*)(ws + 40 * MB);  // 8 MB [M,DIM]

  // Precision staging
  cvt_f32_bf16<<<(kM * kDim / 4 + 255) / 256, 256, 0, stream>>>(x, xbf, kM * kDim / 4);
  dim3 tb(16, 16), tg(kDim / 16, kDim / 16);
  transpose_cvt<<<tg, tb, 0, stream>>>(Wq, WqT);
  transpose_cvt<<<tg, tb, 0, stream>>>(Wk, WkT);
  transpose_cvt<<<tg, tb, 0, stream>>>(Wv, WvT);
  transpose_cvt<<<tg, tb, 0, stream>>>(Wo, WoT);

  // Projections
  dim3 gg(kM / 128, kDim / 128);
  gemm_bf16<0><<<gg, 256, 0, stream>>>(xbf, WqT, bq, Qb);
  gemm_bf16<0><<<gg, 256, 0, stream>>>(xbf, WkT, bk, Kb);
  gemm_bf16<1><<<gg, 256, 0, stream>>>(xbf, WvT, bv, Vtb);

  // Sliding-window attention
  attn_kernel<<<dim3(kSeq / 16, kH, kB), 32, 0, stream>>>(Qb, Kb, Vtb, AObf);

  // Output projection -> f32 d_out
  gemm_bf16<2><<<gg, 256, 0, stream>>>(AObf, WoT, bo, (float*)d_out);
}
